// BoxAlignedMotionEncoder_4818953306582
// MI455X (gfx1250) — matrix operand order repaired and checked by reading, not compiled, not run on hardware
//
#include <hip/hip_runtime.h>
#include <math.h>

#define GRID_SZ 60
#define NCELLS  (GRID_SZ * GRID_SZ)   // 3600
#define N_BOXES 8192
#define FEAT    64
#define TPB     64                    // 2 waves per block
#define BOXES_PER_BLOCK TPB

typedef __attribute__((ext_vector_type(2)))  float    v2f;
typedef __attribute__((ext_vector_type(8)))  float    v8f;
typedef __attribute__((ext_vector_type(16))) _Float16 v16h;

__global__ __launch_bounds__(TPB)
void box_motion_mlp_kernel(const float* __restrict__ mv,     // [2,60,60]
                           const float* __restrict__ boxes,  // [8192,4]
                           const float* __restrict__ W1,     // [8,32]
                           const float* __restrict__ b1,     // [32]
                           const float* __restrict__ W2,     // [32,64]
                           const float* __restrict__ b2,     // [64]
                           float* __restrict__ out)          // [8192,64]
{
    __shared__ float    smv[2 * NCELLS];            // 28.8 KB motion grid
    __shared__ float    sstats[BOXES_PER_BLOCK][8]; // 2 KB stats
    __shared__ _Float16 sh[2][16 * 32];             // 2 KB h staging (per wave)

    const int tid     = threadIdx.x;
    const int boxBase = blockIdx.x * BOXES_PER_BLOCK;

    // ---- stage motion vectors into LDS (fits easily in 320KB WGP LDS) ----
    for (int i = tid; i < 2 * NCELLS; i += TPB) smv[i] = mv[i];
    __syncthreads();

    // ---- phase 1: per-box rectangle reduction (1 lane / box) ----
    {
        const int bi = boxBase + tid;
        float bx1 = boxes[bi * 4 + 0], by1 = boxes[bi * 4 + 1];
        float bx2 = boxes[bi * 4 + 2], by2 = boxes[bi * 4 + 3];
        const float hiC = (float)GRID_SZ - 1e-6f;   // matches jnp.clip(..., GRID-1e-6)
        bx1 = fminf(fmaxf(bx1 * (float)GRID_SZ, 0.f), hiC);
        by1 = fminf(fmaxf(by1 * (float)GRID_SZ, 0.f), hiC);
        bx2 = fminf(fmaxf(bx2 * (float)GRID_SZ, 0.f), hiC);
        by2 = fminf(fmaxf(by2 * (float)GRID_SZ, 0.f), hiC);
        const float fx1 = floorf(bx1), fy1 = floorf(by1);
        const int x1 = (int)fx1, y1 = (int)fy1;
        const int x2 = (int)fmaxf(fx1 + 1.f, ceilf(bx2));
        const int y2 = (int)fmaxf(fy1 + 1.f, ceilf(by2));

        float s0 = 0.f, s1 = 0.f, q0 = 0.f, q1 = 0.f;
        float mn0 = INFINITY, mx0 = -INFINITY, mn1 = INFINITY, mx1 = -INFINITY;
        for (int y = y1; y < y2; ++y) {
            const float* r0 = &smv[y * GRID_SZ];
            const float* r1 = &smv[NCELLS + y * GRID_SZ];
            for (int x = x1; x < x2; ++x) {
                const float v0 = r0[x], v1 = r1[x];
                s0 += v0; q0 += v0 * v0; mn0 = fminf(mn0, v0); mx0 = fmaxf(mx0, v0);
                s1 += v1; q1 += v1 * v1; mn1 = fminf(mn1, v1); mx1 = fmaxf(mx1, v1);
            }
        }
        const float cnt = (float)((x2 - x1) * (y2 - y1));   // >= 1 always
        const float m0 = s0 / cnt, m1 = s1 / cnt;
        const float ss0 = fmaxf(q0 - s0 * m0, 0.f);         // sum((v-mean)^2)
        const float ss1 = fmaxf(q1 - s1 * m1, 0.f);
        const float sd0 = (cnt > 1.f) ? sqrtf(ss0 / (cnt - 1.f)) : 0.f;
        const float sd1 = (cnt > 1.f) ? sqrtf(ss1 / (cnt - 1.f)) : 0.f;
        sstats[tid][0] = m0;  sstats[tid][1] = m1;
        sstats[tid][2] = sd0; sstats[tid][3] = sd1;
        sstats[tid][4] = mn0; sstats[tid][5] = mx0;
        sstats[tid][6] = mn1; sstats[tid][7] = mx1;
    }
    __syncthreads();

    // ---- phase 2: MLP via WMMA (wave32, 16-row tiles) ----
    const int wave = tid >> 5;
    const int lane = tid & 31;
    const int lr   = lane & 15;   // row/col index within a 16-tile
    const int hi   = lane >> 4;   // lane-half select per ISA layouts

    for (int s = 0; s < 2; ++s) {               // 2 tiles per wave
        const int tileRow = (wave * 2 + s) * 16;   // local row base (0..48)

        // ----- layer 1: stats[16x8] @ W1[8x32] as two chained f32 16x16x4 WMMAs -----
        v8f c1[2];
        #pragma unroll
        for (int nt = 0; nt < 2; ++nt) {
            const float bias = b1[nt * 16 + lr];   // bias depends on N only
            #pragma unroll
            for (int j = 0; j < 8; ++j) c1[nt][j] = bias;
        }
        #pragma unroll
        for (int c = 0; c < 2; ++c) {           // K chunks of 4
            v2f a;                              // A 16x4 f32: lane=M, elem e -> K = c*4 + 2*hi + e
            a[0] = sstats[tileRow + lr][c * 4 + 2 * hi + 0];
            a[1] = sstats[tileRow + lr][c * 4 + 2 * hi + 1];
            #pragma unroll
            for (int nt = 0; nt < 2; ++nt) {
                v2f b;                          // B 4x16 f32: lane=N, elem e -> K = c*4 + 2*hi + e
                const int col = nt * 16 + lr;
                b[0] = W1[(c * 4 + 2 * hi + 0) * 32 + col];
                b[1] = W1[(c * 4 + 2 * hi + 1) * 32 + col];
                c1[nt] = __builtin_amdgcn_wmma_f32_16x16x4_f32(
                    false, a, false, b, (short)0, c1[nt], false, false);
            }
        }

        // ----- relu, stage h (16x32) as f16 via per-wave LDS (layout shuffle C->A) -----
        #pragma unroll
        for (int nt = 0; nt < 2; ++nt)
            #pragma unroll
            for (int j = 0; j < 8; ++j) {
                const int m   = j + 8 * hi;     // C layout: elem j -> M = j + 8*hi
                const int col = nt * 16 + lr;   //           N = lane&15 (+tile)
                sh[wave][m * 32 + col] = (_Float16)fmaxf(c1[nt][j], 0.f);
            }
        // same-wave LDS RAW: DS ops are in-order per wave; compiler inserts s_wait_dscnt

        // A 16x32 f16 (ISA 7.12.2): elem p<8 -> K = 8*hi + p ; p>=8 -> K = 16 + 8*hi + (p-8)
        v16h a2;
        #pragma unroll
        for (int p = 0; p < 8; ++p) a2[p]     = sh[wave][lr * 32 + 8 * hi + p];
        #pragma unroll
        for (int p = 0; p < 8; ++p) a2[8 + p] = sh[wave][lr * 32 + 16 + 8 * hi + p];

        // ----- layer 2: h[16x32] @ W2[32x64] as one f16 16x16x32 WMMA per 16-col tile -----
        const int rowBase = boxBase + tileRow;
        #pragma unroll
        for (int nt = 0; nt < 4; ++nt) {
            const int col = nt * 16 + lr;
            v16h bm;                            // B 32x16 f16: elem p<8 -> K = 8*hi + p ; p>=8 -> K = 16 + 8*hi + (p-8), N = lane&15
            #pragma unroll
            for (int p = 0; p < 16; ++p)
                bm[p] = (_Float16)W2[(8 * hi + p + (p >= 8 ? 8 : 0)) * FEAT + col];
            v8f c2;
            const float bias = b2[col];
            #pragma unroll
            for (int j = 0; j < 8; ++j) c2[j] = bias;
            c2 = __builtin_amdgcn_wmma_f32_16x16x32_f16(
                false, a2, false, bm, (short)0, c2, false, false);
            #pragma unroll
            for (int j = 0; j < 8; ++j)
                out[(rowBase + j + 8 * hi) * FEAT + col] = c2[j];
        }
    }
}

extern "C" void kernel_launch(void* const* d_in, const int* in_sizes, int n_in,
                              void* d_out, int out_size, void* d_ws, size_t ws_size,
                              hipStream_t stream) {
    const float* mv    = (const float*)d_in[0];
    const float* boxes = (const float*)d_in[1];
    const float* W1    = (const float*)d_in[2];
    const float* b1    = (const float*)d_in[3];
    const float* W2    = (const float*)d_in[4];
    const float* b2    = (const float*)d_in[5];
    float* out = (float*)d_out;

    dim3 grid(N_BOXES / BOXES_PER_BLOCK);   // 128 blocks
    dim3 block(TPB);                        // 64 threads = 2 wave32
    hipLaunchKernelGGL(box_motion_mlp_kernel, grid, block, 0, stream,
                       mv, boxes, W1, b1, W2, b2, out);
}
